// TriplaneSynthesizer_25632364822829
// MI455X (gfx1250) — matrix operand order repaired and checked by reading, not compiled, not run on hardware
//
#include <hip/hip_runtime.h>
#include <hip/hip_bf16.h>

// Triplane renderer for MI455X (gfx1250, wave32).
// Layer-1 MLP (240->64 over 1.57M points) runs on v_wmma_f32_16x16x32_f16.
// Per-wave tiles are fully decoupled: only one block barrier (W1 staging);
// intra-wave LDS RAW/WAR relies on CDNA5's in-order LDS pipeline + a
// compiler scheduling fence (__builtin_amdgcn_wave_barrier).

typedef _Float16 half_t;
typedef __attribute__((ext_vector_type(16))) _Float16 v16h;
typedef __attribute__((ext_vector_type(8)))  float    v8f;

#define RS      64
#define RAYS    (RS * RS)          // 4096
#define SPR     96
#define MPTS    (RAYS * SPR)       // 393216
#define KDIM    240
#define KPAD    256
#define HID     64
#define BOXMIN  (-0.6f)
#define BOXMAX  (0.6f)
#define EPSFD   0.01f
#define WPITCH  264                // halves per row of staged W1^T (pad vs bank conflicts)
#define APITCH  40                 // halves per row of per-wave A tile (32 + pad)
#define WAVES   8
#define TILES_PER_VARIANT (MPTS / 16)   // 24576
#define TOTAL_TILES (4 * TILES_PER_VARIANT)

__global__ __launch_bounds__(256) void decode_kernel(
    const float* __restrict__ planes, const float* __restrict__ ro_g,
    const float* __restrict__ rd_g,  const float* __restrict__ W1,
    const float* __restrict__ b1,    const float* __restrict__ W2,
    const float* __restrict__ b2,    float* __restrict__ dec)
{
    __shared__ half_t sW1t[HID * WPITCH];        // W1 transposed [n][k], f16, k padded to 256
    __shared__ half_t sA[WAVES * 16 * APITCH];   // per-wave 16x32 feature tile (f16)
    // Per (row, plane): 4 clamped tap offsets (as int bits) + 4 validity-masked
    // bilinear weights. Row stride 32 dwords (3 planes x 8, padded).
    __shared__ float  sTap[WAVES * 16 * 32];

    const int tid     = threadIdx.x;
    const int lane    = tid & 31;
    const int wave    = tid >> 5;
    const int half_id = (lane >> 4) & 1;
    const int l16     = lane & 15;

    // ---- Stage W1^T into LDS as f16 (zero-padded K) ----
    for (int idx = tid; idx < HID * KPAD; idx += 256) {
        int k = idx >> 6;        // 0..255
        int n = idx & 63;
        float w = (k < KDIM) ? W1[k * HID + n] : 0.0f;
        sW1t[n * WPITCH + k] = (half_t)w;
    }

    const int tile    = blockIdx.x * WAVES + wave;
    const int gp0     = tile * 16;
    const int variant = tile / TILES_PER_VARIANT;   // 0 = base, 1..3 = FD offsets

    // ---- Per-row point projection + bilinear tap precompute (lanes 0..15) ----
    if (lane < 16) {
        int p   = (gp0 % MPTS) + l16;
        int ray = p / SPR;
        int s   = p - ray * SPR;
        float rox = ro_g[ray * 3 + 0], roy = ro_g[ray * 3 + 1], roz = ro_g[ray * 3 + 2];
        float rdx = rd_g[ray * 3 + 0], rdy = rd_g[ray * 3 + 1], rdz = rd_g[ray * 3 + 2];
        float sx = (fabsf(rdx) < 1e-9f) ? 1e-9f : rdx;
        float sy = (fabsf(rdy) < 1e-9f) ? 1e-9f : rdy;
        float sz = (fabsf(rdz) < 1e-9f) ? 1e-9f : rdz;
        float t1x = (BOXMIN - rox) / sx, t2x = (BOXMAX - rox) / sx;
        float t1y = (BOXMIN - roy) / sy, t2y = (BOXMAX - roy) / sy;
        float t1z = (BOXMIN - roz) / sz, t2z = (BOXMAX - roz) / sz;
        float tn = fmaxf(fmaxf(fminf(t1x, t2x), fminf(t1y, t2y)), fminf(t1z, t2z));
        tn = fmaxf(tn, 0.0f);
        float tf = fminf(fminf(fmaxf(t1x, t2x), fmaxf(t1y, t2y)), fmaxf(t1z, t2z));
        float span = fmaxf(tf - tn, 0.0f);
        float depth = tn + span * (((float)s + 0.5f) / (float)SPR);
        float px = rox + depth * rdx;
        float py = roy + depth * rdy;
        float pz = roz + depth * rdz;
        if (variant == 1) px += EPSFD;
        if (variant == 2) py += EPSFD;
        if (variant == 3) pz += EPSFD;
        if (variant != 0) {
            px = fminf(fmaxf(px, BOXMIN), BOXMAX);
            py = fminf(fmaxf(py, BOXMIN), BOXMAX);
            pz = fminf(fmaxf(pz, BOXMIN), BOXMAX);
        }
        const float sc = 2.0f / 1.2f;
        float nx = px * sc, ny = py * sc, nz = pz * sc;
        // projections: plane0 (x,y), plane1 (x,z), plane2 (y,z)
        float us[3] = {nx, nx, ny};
        float vs[3] = {ny, nz, nz};
        float* tp = &sTap[(wave * 16 + l16) * 32];
#pragma unroll
        for (int pl = 0; pl < 3; ++pl) {
            // grid_sample coord: x = ((u+1)*64 - 1)*0.5 = (u+1)*32 - 0.5
            float xf = (us[pl] + 1.0f) * 32.0f - 0.5f;
            float yf = (vs[pl] + 1.0f) * 32.0f - 0.5f;
            float x0f = floorf(xf), y0f = floorf(yf);
            int x0 = (int)x0f, y0 = (int)y0f;
            float wx = xf - x0f, wy = yf - y0f;
            int x0c = min(max(x0, 0), RS - 1),     y0c = min(max(y0, 0), RS - 1);
            int x1c = min(max(x0 + 1, 0), RS - 1), y1c = min(max(y0 + 1, 0), RS - 1);
            float vx0 = (x0 >= 0 && x0 < RS) ? 1.0f : 0.0f;
            float vx1 = (x0 + 1 >= 0 && x0 + 1 < RS) ? 1.0f : 0.0f;
            float vy0 = (y0 >= 0 && y0 < RS) ? 1.0f : 0.0f;
            float vy1 = (y0 + 1 >= 0 && y0 + 1 < RS) ? 1.0f : 0.0f;
            tp[pl * 8 + 0] = __int_as_float(y0c * RS + x0c);
            tp[pl * 8 + 1] = __int_as_float(y0c * RS + x1c);
            tp[pl * 8 + 2] = __int_as_float(y1c * RS + x0c);
            tp[pl * 8 + 3] = __int_as_float(y1c * RS + x1c);
            tp[pl * 8 + 4] = (1.f - wx) * (1.f - wy) * vx0 * vy0;
            tp[pl * 8 + 5] = wx * (1.f - wy) * vx1 * vy0;
            tp[pl * 8 + 6] = (1.f - wx) * wy * vx0 * vy1;
            tp[pl * 8 + 7] = wx * wy * vx1 * vy1;
        }
    }
    __syncthreads();   // covers sW1t staging (block-wide) and per-wave sTap

    v8f acc[4];
    const v8f vzero = {0.f, 0.f, 0.f, 0.f, 0.f, 0.f, 0.f, 0.f};
#pragma unroll
    for (int nt = 0; nt < 4; ++nt) acc[nt] = vzero;

#pragma unroll 1
    for (int kk = 0; kk < 8; ++kk) {
        // ---- Feature fill: lane handles feature column c for all 16 rows ----
        // Branch-free: K-pad lanes (c >= 240) gather a clamped valid channel
        // and multiply by 0 (exactly zero contribution).
        int c = kk * 32 + lane;
        float cmask = (c < KDIM) ? 1.0f : 0.0f;
        int pl = c / 80; if (pl > 2) pl = 2;
        int ch = c - pl * 80; if (ch > 79) ch = 79;
        const float* fbase = planes + (size_t)(pl * 80 + ch) * (RS * RS);
        const int plo = pl * 8;
#pragma unroll 1
        for (int r = 0; r < 16; ++r) {
            const float* tp = &sTap[(wave * 16 + r) * 32 + plo];
            int   o0 = __float_as_int(tp[0]), o1 = __float_as_int(tp[1]);
            int   o2 = __float_as_int(tp[2]), o3 = __float_as_int(tp[3]);
            float w0 = tp[4], w1 = tp[5], w2 = tp[6], w3 = tp[7];
            float val = (fbase[o0] * w0 + fbase[o1] * w1
                       + fbase[o2] * w2 + fbase[o3] * w3) * cmask;
            sA[(wave * 16 + r) * APITCH + lane] = (half_t)val;
        }
        __builtin_amdgcn_wave_barrier();   // fence: fill stores before fragment loads
                                           // (CDNA5 LDS is in-order within a wave)

        // ---- A fragment: 16-bit A 16x32 layout (lanes 0-15: V0-3 K0-7, V4-7 K16-23; +8 for hi half)
        v16h a;
        const half_t* rowp = &sA[(wave * 16 + l16) * APITCH];
#pragma unroll
        for (int i = 0; i < 8; ++i) {
            int kl = ((i < 4) ? (2 * i) : (8 + 2 * i)) + 8 * half_id;
            a[2 * i]     = rowp[kl];
            a[2 * i + 1] = rowp[kl + 1];
        }
        // ---- B fragments: column n = nt*16 + lane%16, k = kk*32 + 8*half + (i<4 ? 2i : 8+2i) (+1)
#pragma unroll
        for (int nt = 0; nt < 4; ++nt) {
            v16h b;
            const half_t* wp = &sW1t[(nt * 16 + l16) * WPITCH + kk * 32 + 8 * half_id];
#pragma unroll
            for (int i = 0; i < 8; ++i) {
                int kb = (i < 4) ? (2 * i) : (8 + 2 * i);
                b[2 * i]     = wp[kb];
                b[2 * i + 1] = wp[kb + 1];
            }
            acc[nt] = __builtin_amdgcn_wmma_f32_16x16x32_f16(
                false, a, false, b, (short)0, acc[nt], false, false);
        }
        __builtin_amdgcn_wave_barrier();   // fence: fragment loads before next fill stores
    }

    // ---- Epilogue: bias + ReLU, then 64->4 projection with xor-butterfly reduce ----
    float b1v[4], w2v[4][4];
#pragma unroll
    for (int nt = 0; nt < 4; ++nt) {
        int n = nt * 16 + l16;
        b1v[nt] = b1[n];
#pragma unroll
        for (int j = 0; j < 4; ++j) w2v[nt][j] = W2[n * 4 + j];
    }
    float outv[8][4];
#pragma unroll
    for (int v = 0; v < 8; ++v) {
        float s0 = 0.f, s1 = 0.f, s2 = 0.f, s3 = 0.f;
#pragma unroll
        for (int nt = 0; nt < 4; ++nt) {
            float h = fmaxf(acc[nt][v] + b1v[nt], 0.0f);
            s0 += h * w2v[nt][0]; s1 += h * w2v[nt][1];
            s2 += h * w2v[nt][2]; s3 += h * w2v[nt][3];
        }
        outv[v][0] = s0; outv[v][1] = s1; outv[v][2] = s2; outv[v][3] = s3;
    }
#pragma unroll
    for (int v = 0; v < 8; ++v)
#pragma unroll
        for (int j = 0; j < 4; ++j) {
            float x = outv[v][j];
            x += __shfl_xor(x, 1, 32);
            x += __shfl_xor(x, 2, 32);
            x += __shfl_xor(x, 4, 32);
            x += __shfl_xor(x, 8, 32);
            outv[v][j] = x;
        }
    if (l16 == 0) {
        int mbase = gp0 + 8 * half_id;   // row M = v + 8*half
#pragma unroll
        for (int v = 0; v < 8; ++v)
#pragma unroll
            for (int j = 0; j < 4; ++j)
                dec[(size_t)(mbase + v) * 4 + j] = outv[v][j] + b2[j];
    }
}

__global__ __launch_bounds__(256) void composite_kernel(
    const float* __restrict__ ro_g, const float* __restrict__ rd_g,
    const float* __restrict__ dec,  float* __restrict__ out)
{
    int r = blockIdx.x * blockDim.x + threadIdx.x;
    if (r >= RAYS) return;
    float rox = ro_g[r * 3 + 0], roy = ro_g[r * 3 + 1], roz = ro_g[r * 3 + 2];
    float rdx = rd_g[r * 3 + 0], rdy = rd_g[r * 3 + 1], rdz = rd_g[r * 3 + 2];
    float sx = (fabsf(rdx) < 1e-9f) ? 1e-9f : rdx;
    float sy = (fabsf(rdy) < 1e-9f) ? 1e-9f : rdy;
    float sz = (fabsf(rdz) < 1e-9f) ? 1e-9f : rdz;
    float t1x = (BOXMIN - rox) / sx, t2x = (BOXMAX - rox) / sx;
    float t1y = (BOXMIN - roy) / sy, t2y = (BOXMAX - roy) / sy;
    float t1z = (BOXMIN - roz) / sz, t2z = (BOXMAX - roz) / sz;
    float tn = fmaxf(fmaxf(fminf(t1x, t2x), fminf(t1y, t2y)), fminf(t1z, t2z));
    tn = fmaxf(tn, 0.0f);
    float tf = fminf(fminf(fmaxf(t1x, t2x), fmaxf(t1y, t2y)), fmaxf(t1z, t2z));
    float span  = fmaxf(tf - tn, 0.0f);
    float delta = span / (float)SPR;

    float T = 1.0f, wsum = 0.f;
    float rgb0 = 0.f, rgb1 = 0.f, rgb2 = 0.f, dsum = 0.f;
    float nxs = 0.f, nys = 0.f, nzs = 0.f;

    for (int s = 0; s < SPR; ++s) {
        size_t p = (size_t)r * SPR + s;
        float sdf  = dec[p * 4 + 0];
        float raw1 = dec[p * 4 + 1];
        float raw2 = dec[p * 4 + 2];
        float raw3 = dec[p * 4 + 3];
        float g0 = (dec[((size_t)MPTS * 1 + p) * 4] - sdf) * (1.0f / EPSFD);
        float g1 = (dec[((size_t)MPTS * 2 + p) * 4] - sdf) * (1.0f / EPSFD);
        float g2 = (dec[((size_t)MPTS * 3 + p) * 4] - sdf) * (1.0f / EPSFD);
        out[32768 + p * 3 + 0] = g0;   // sdf_grad output
        out[32768 + p * 3 + 1] = g1;
        out[32768 + p * 3 + 2] = g2;
        float gn = sqrtf(g0 * g0 + g1 * g1 + g2 * g2) + 1e-8f;
        float n0 = g0 / gn, n1 = g1 / gn, n2 = g2 / gn;

        float depth = tn + span * (((float)s + 0.5f) / (float)SPR);
        float px = rox + depth * rdx;
        float py = roy + depth * rdy;
        float pz = roz + depth * rdz;
        float pn = sqrtf(px * px + py * py + pz * pz);
        float shifted = sdf + (pn - 0.5f);
        float xarg = -shifted * 80.0f;
        float sigma = fmaxf(xarg, 0.0f) + log1pf(expf(-fabsf(xarg)));   // stable softplus
        float alpha = 1.0f - expf(-sigma * delta);
        float w = alpha * T;
        T *= (1.0f - alpha + 1e-10f);

        float c0 = 1.002f / (1.0f + expf(-raw1)) - 0.001f;
        float c1 = 1.002f / (1.0f + expf(-raw2)) - 0.001f;
        float c2 = 1.002f / (1.0f + expf(-raw3)) - 0.001f;
        rgb0 += w * c0; rgb1 += w * c1; rgb2 += w * c2;
        dsum += w * depth; wsum += w;
        nxs += w * n0; nys += w * n1; nzs += w * n2;
    }
    float bg = 1.0f - wsum;
    rgb0 += bg; rgb1 += bg; rgb2 += bg;
    float nn = sqrtf(nxs * nxs + nys * nys + nzs * nzs) + 1e-8f;
    float m0 = ((nxs / nn) + 1.0f) * 0.5f * wsum;
    float m1 = ((nys / nn) + 1.0f) * 0.5f * wsum;
    float m2 = ((nzs / nn) + 1.0f) * 0.5f * wsum;

    out[0 * RAYS + r] = rgb0;                // comp_rgb (1,3,64,64)
    out[1 * RAYS + r] = rgb1;
    out[2 * RAYS + r] = rgb2;
    out[12288 + r]    = dsum;                // comp_depth
    out[16384 + r]    = wsum;                // opacity
    out[20480 + 0 * RAYS + r] = m0;          // comp_normal
    out[20480 + 1 * RAYS + r] = m1;
    out[20480 + 2 * RAYS + r] = m2;
}

extern "C" void kernel_launch(void* const* d_in, const int* in_sizes, int n_in,
                              void* d_out, int out_size, void* d_ws, size_t ws_size,
                              hipStream_t stream) {
    (void)in_sizes; (void)n_in; (void)out_size; (void)ws_size;
    const float* planes = (const float*)d_in[0];
    const float* ro     = (const float*)d_in[1];
    const float* rd     = (const float*)d_in[2];
    const float* W1     = (const float*)d_in[3];
    const float* b1     = (const float*)d_in[4];
    const float* W2     = (const float*)d_in[5];
    const float* b2     = (const float*)d_in[6];
    float* dec = (float*)d_ws;                 // 4 * MPTS * 4 floats (~24 MB)
    float* out = (float*)d_out;

    dim3 block(256);
    dim3 grid1(TOTAL_TILES / WAVES);           // 98304 tiles / 8 waves = 12288 blocks
    decode_kernel<<<grid1, block, 0, stream>>>(planes, ro, rd, W1, b1, W2, b2, dec);
    composite_kernel<<<(RAYS + 255) / 256, block, 0, stream>>>(ro, rd, dec, out);
}
